// BertAdapterCapsuleMask_51840255263220
// MI455X (gfx1250) — matrix operand order repaired and checked by reading, not compiled, not run on hardware
//
#include <hip/hip_runtime.h>

// ---------------------------------------------------------------------------
// Types for CDNA5 WMMA
// ---------------------------------------------------------------------------
typedef __attribute__((ext_vector_type(16))) __bf16 v16bf;
typedef __attribute__((ext_vector_type(8)))  float  v8f;

union Frag16 { v16bf v; uint4 q[2]; };

__device__ __forceinline__ unsigned short f2bf(float f) {
    unsigned int u = __builtin_bit_cast(unsigned int, f);
    u += 0x7FFFu + ((u >> 16) & 1u);            // round-to-nearest-even
    return (unsigned short)(u >> 16);
}

// ---------------------------------------------------------------------------
// Generic bf16 WMMA GEMM core:  out[M,N] = epilogue( A[M,K] * W[N,K]^T )
// Block tile 128(M) x 128(N), BK=32, 256 threads = 8 wave32 (2 Mgrp x 4 Ngrp),
// each wave computes a 64x32 tile = 4x2 v_wmma_f32_16x16x32_bf16 fragments.
// LDS is double-buffered (ping-pong): one barrier per K-step, global loads for
// tile k+1 issued before the WMMAs of tile k so XDL work hides HBM latency.
// Epilogue configuration is compile-time (no runtime-null branches => no
// accumulator spills to scratch).
// ---------------------------------------------------------------------------
template <bool HB, bool HG, bool RELU, bool HR, bool OBF>
__device__ __forceinline__ void gemm_core(
    const unsigned short* __restrict__ A, int lda,
    const unsigned short* __restrict__ W, int ldw,
    int row0, int col0, int K,
    float* __restrict__ outF, unsigned short* __restrict__ outBF,
    int ldc, int Nreal,
    const float* __restrict__ bias, const float* __restrict__ gate,
    const float* __restrict__ resid)
{
    __shared__ __align__(16) unsigned short As[2][128][40];  // +8 pad, 16B rows
    __shared__ __align__(16) unsigned short Bs[2][128][40];

    const int tid  = threadIdx.x;
    const int lane = tid & 31;
    const int wid  = tid >> 5;
    const int mw   = wid >> 2;   // 0..1 : 64-row group
    const int nw   = wid & 3;    // 0..3 : 32-col group

    // per-thread staging coords: 16 bf16 of A and 16 bf16 of B per K-step
    const int sr = tid >> 1;            // 0..127
    const int sk = (tid & 1) * 16;      // 0 or 16
    const unsigned short* srcA = A + (size_t)(row0 + sr) * lda + sk;
    const unsigned short* srcB = W + (size_t)(col0 + sr) * ldw + sk;

    v8f acc[4][2];
    for (int i = 0; i < 4; i++)
        for (int j = 0; j < 2; j++)
            for (int r = 0; r < 8; r++) acc[i][j][r] = 0.0f;

    // prologue: stage K-tile 0 into buffer 0
    {
        uint4 a0 = *(const uint4*)(srcA);
        uint4 a1 = *(const uint4*)(srcA + 8);
        uint4 b0 = *(const uint4*)(srcB);
        uint4 b1 = *(const uint4*)(srcB + 8);
        *(uint4*)&As[0][sr][sk]     = a0;
        *(uint4*)&As[0][sr][sk + 8] = a1;
        *(uint4*)&Bs[0][sr][sk]     = b0;
        *(uint4*)&Bs[0][sr][sk + 8] = b1;
    }
    __syncthreads();

    const int nk = K >> 5;
    for (int kt = 0; kt < nk; kt++) {
        const int cur = kt & 1;
        const bool pref = (kt + 1) < nk;
        uint4 a0, a1, b0, b1;
        if (pref) {                       // issue global loads for tile kt+1
            const int k0 = (kt + 1) * 32;
            a0 = *(const uint4*)(srcA + k0);
            a1 = *(const uint4*)(srcA + k0 + 8);
            b0 = *(const uint4*)(srcB + k0);
            b1 = *(const uint4*)(srcB + k0 + 8);
        }

        // fragment loads from current LDS buffer + 8 WMMAs
        Frag16 af[4], bf[2];
        const int kga = (lane >> 4) * 8;     // A: k chunks at kga and kga+16
        for (int mt = 0; mt < 4; mt++) {
            const int rr = mw * 64 + mt * 16 + (lane & 15);
            af[mt].q[0] = *(const uint4*)&As[cur][rr][kga];
            af[mt].q[1] = *(const uint4*)&As[cur][rr][kga + 16];
        }
        const int kgb = (lane >> 4) * 8;     // B: k chunks at kgb and kgb+16
        for (int nt = 0; nt < 2; nt++) {
            const int cc = nw * 32 + nt * 16 + (lane & 15);
            bf[nt].q[0] = *(const uint4*)&Bs[cur][cc][kgb];
            bf[nt].q[1] = *(const uint4*)&Bs[cur][cc][kgb + 16];
        }
        for (int mt = 0; mt < 4; mt++)
            for (int nt = 0; nt < 2; nt++)
                acc[mt][nt] = __builtin_amdgcn_wmma_f32_16x16x32_bf16(
                    false, af[mt].v, false, bf[nt].v,
                    (short)0, acc[mt][nt], false, false);

        if (pref) {                       // store staged tile into other buffer
            const int nxt = cur ^ 1;
            *(uint4*)&As[nxt][sr][sk]     = a0;
            *(uint4*)&As[nxt][sr][sk + 8] = a1;
            *(uint4*)&Bs[nxt][sr][sk]     = b0;
            *(uint4*)&Bs[nxt][sr][sk + 8] = b1;
        }
        __syncthreads();                  // single barrier per K-step
    }

    // Epilogue: D layout — VGPR r holds row (r + 8*(lane>>4)), col = lane&15
    for (int mt = 0; mt < 4; mt++)
        for (int nt = 0; nt < 2; nt++) {
            const int colg = col0 + nw * 32 + nt * 16 + (lane & 15);
            if (colg < Nreal) {
                const float b = HB ? bias[colg] : 0.0f;
                const float g = HG ? gate[colg] : 1.0f;
                for (int r = 0; r < 8; r++) {
                    const int rowg = row0 + mw * 64 + mt * 16 + r + 8 * (lane >> 4);
                    float v = acc[mt][nt][r];
                    if (HB)   v += b;
                    if (RELU) v = fmaxf(v, 0.0f);
                    if (HG)   v *= g;
                    const size_t o = (size_t)rowg * ldc + colg;
                    if (HR)   v += resid[o];
                    if (OBF)  outBF[o] = f2bf(v);
                    else      outF[o]  = v;
                }
            }
        }
}

template <bool HB, bool HG, bool RELU, bool HR, bool OBF>
__global__ __launch_bounds__(256) void gemm_kernel(
    const unsigned short* A, int lda, const unsigned short* W, int ldw, int K,
    float* outF, unsigned short* outBF, int ldc, int Nreal,
    const float* bias, const float* gate, const float* resid)
{
    gemm_core<HB, HG, RELU, HR, OBF>(
        A, lda, W, ldw, blockIdx.y * 128, blockIdx.x * 128, K,
        outF, outBF, ldc, Nreal, bias, gate, resid);
}

// priors[c,b,n,l] = sem_sq[b,n,:] (128x384) @ RwT[c,n] (128x384)^T ; batch z = c*10+n
__global__ __launch_bounds__(256) void priors_kernel(
    const unsigned short* semsq, const unsigned short* rwt, float* priors)
{
    const int z = blockIdx.y;
    const int c = z / 10, n = z % 10;
    gemm_core<false, false, false, false, false>(
        semsq + n * 384, 3840,
        rwt + (size_t)z * 128 * 384, 384,
        0, 0, 384,
        priors + (size_t)c * 163840 + n * 128, nullptr,
        1280, 128, nullptr, nullptr, nullptr);
}

// ---------------------------------------------------------------------------
// Prep kernels
// ---------------------------------------------------------------------------
__global__ void cvt_bf16_kernel(const float* __restrict__ src,
                                unsigned short* __restrict__ dst, long n)
{
    long i = (long)blockIdx.x * blockDim.x + threadIdx.x;
    const long stride = (long)gridDim.x * blockDim.x;
    for (; i < n; i += stride) dst[i] = f2bf(src[i]);
}

// RwT[(c*10+n)][l][d] = route_weights[c,n,d,l]  (bf16)
__global__ void transpose_rw_kernel(const float* __restrict__ rw,
                                    unsigned short* __restrict__ rwt)
{
    const long total = 30L * 128 * 384;
    long i = (long)blockIdx.x * blockDim.x + threadIdx.x;
    const long stride = (long)gridDim.x * blockDim.x;
    for (; i < total; i += stride) {
        const long cn = i / (128 * 384);
        const long rem = i % (128 * 384);
        const long l = rem / 384;
        const long d = rem % 384;
        rwt[i] = f2bf(rw[(cn * 384 + d) * 128 + l]);
    }
}

// Fuse the two (no-activation) capsule linear layers:
// Wc[j=n*3+c, h] = sum_k sfc2_w[n,c,k]*sfc1_w[n,k,h];  bc[j] = sfc2_b + sfc2_w·sfc1_b
__global__ void fuse_wc_kernel(const float* __restrict__ sfc1_w,
                               const float* __restrict__ sfc1_b,
                               const float* __restrict__ sfc2_w,
                               const float* __restrict__ sfc2_b,
                               unsigned short* __restrict__ wc,
                               float* __restrict__ bc)
{
    const int idx = blockIdx.x * blockDim.x + threadIdx.x;
    if (idx >= 128 * 768) return;
    const int j = idx / 768, h = idx % 768;
    float acc = 0.0f;
    if (j < 30) {
        const int n = j / 3, c = j % 3;
        const float* w2 = sfc2_w + (n * 3 + c) * 100;
        for (int k = 0; k < 100; k++)
            acc += w2[k] * sfc1_w[((size_t)n * 100 + k) * 768 + h];
    }
    wc[(size_t)j * 768 + h] = f2bf(acc);
    if (h == 0) {
        float b = 0.0f;
        if (j < 30) {
            const int n = j / 3, c = j % 3;
            const float* w2 = sfc2_w + (n * 3 + c) * 100;
            b = sfc2_b[n * 3 + c];
            for (int k = 0; k < 100; k++) b += w2[k] * sfc1_b[n * 100 + k];
        }
        bc[j] = b;
    }
}

__global__ void gates_kernel(const int* __restrict__ tptr, const float* __restrict__ sptr,
                             const float* __restrict__ efc1, const float* __restrict__ efc2,
                             const float* __restrict__ elarger,
                             float* __restrict__ gfc1, float* __restrict__ gfc2,
                             float* __restrict__ glg)
{
    const int i = blockIdx.x * blockDim.x + threadIdx.x;
    const int t = *tptr;
    const float s = *sptr;
    if (i < 2048) gfc1[i] = 1.0f / (1.0f + __expf(-s * efc1[(size_t)t * 2048 + i]));
    if (i < 768) {
        gfc2[i] = 1.0f / (1.0f + __expf(-s * efc2[(size_t)t * 768 + i]));
        glg[i]  = 1.0f / (1.0f + __expf(-s * elarger[(size_t)t * 768 + i]));
    }
}

// squash over the task axis: sem_raw[(b*128+l)*32 + n*3+c] -> sem_sq[b][n][l*3+c] (bf16)
__global__ void squash_tasks_kernel(const float* __restrict__ sem_raw,
                                    unsigned short* __restrict__ sem_sq)
{
    const int idx = blockIdx.x * blockDim.x + threadIdx.x;
    if (idx >= 128 * 128 * 3) return;
    const int c = idx % 3;
    const int l = (idx / 3) % 128;
    const int b = idx / (3 * 128);
    float v[10], sq = 0.0f;
    const float* row = sem_raw + ((size_t)b * 128 + l) * 32;
    for (int n = 0; n < 10; n++) { v[n] = row[n * 3 + c]; sq += v[n] * v[n]; }
    const float scale = sqrtf(sq) / (1.0f + sq);   // (sq/(1+sq))/sqrt(sq)
    for (int n = 0; n < 10; n++)
        sem_sq[((size_t)b * 10 + n) * 384 + l * 3 + c] = f2bf(v[n] * scale);
}

// ---------------------------------------------------------------------------
// Dynamic routing: one 128-thread block per (c,b); thread tid owns l.
// ---------------------------------------------------------------------------
__device__ __forceinline__ float block_reduce_sum(float v, float* red)
{
    const int tid = threadIdx.x;
    red[tid] = v;
    __syncthreads();
    for (int s = 64; s > 0; s >>= 1) {
        if (tid < s) red[tid] += red[tid + s];
        __syncthreads();
    }
    const float r = red[0];
    __syncthreads();
    return r;
}

__global__ __launch_bounds__(128) void routing_kernel(const float* __restrict__ priors,
                                                      const int* __restrict__ tptr,
                                                      float* __restrict__ vote)
{
    __shared__ float red[128];
    const int bx = blockIdx.x;          // c*128 + b
    const int l = threadIdx.x;
    const int t = *tptr;
    float pr[10];
    for (int n = 0; n < 10; n++)
        pr[n] = priors[((size_t)bx * 10 + n) * 128 + l];

    float logits[10];
    for (int n = 0; n < 10; n++) logits[n] = 0.0f;

    float voteV = 0.0f;
    for (int it = 0; it < 3; it++) {
        float lm[10], mx = -1e30f;
        for (int n = 0; n < 10; n++) {
            lm[n] = (n <= t) ? logits[n] : -10000.0f;
            mx = fmaxf(mx, lm[n]);
        }
        float p[10], se = 0.0f;
        for (int n = 0; n < 10; n++) { p[n] = __expf(lm[n] - mx); se += p[n]; }
        const float inv = 1.0f / se;
        voteV = 0.0f;
        for (int n = 0; n < 10; n++) { p[n] *= inv; voteV += p[n] * pr[n]; }
        if (it < 2) {
            const float sq = block_reduce_sum(voteV * voteV, red);   // squash over l
            const float outl = voteV * sqrtf(sq) / (1.0f + sq);
            for (int n = 0; n < 10; n++) {
                const float d = block_reduce_sum(pr[n] * outl, red); // uniform over l
                logits[n] = lm[n] + d;
            }
        }
    }
    vote[(size_t)bx * 128 + l] = voteV;
}

// ---------------------------------------------------------------------------
// h = bf16( x + ((h_out @ larger_w^T + larger_b) * glarger) ); one block per token
// h_out uses the torch .view scramble: value = vote_flat[b*384 + l*3 + c]
// ---------------------------------------------------------------------------
__global__ __launch_bounds__(256) void build_h_kernel(const float* __restrict__ x,
                                                      const float* __restrict__ vote,
                                                      const float* __restrict__ larger_w,
                                                      const float* __restrict__ larger_b,
                                                      const float* __restrict__ glg,
                                                      unsigned short* __restrict__ hbf)
{
    __shared__ float hv[3];
    const int row = blockIdx.x;         // b*128 + l
    const int b = row >> 7, l = row & 127;
    if (threadIdx.x < 3)
        hv[threadIdx.x] = vote[(size_t)b * 384 + l * 3 + threadIdx.x];
    __syncthreads();
    for (int hid = threadIdx.x; hid < 768; hid += 256) {
        float cap = larger_b[hid];
        cap += hv[0] * larger_w[hid * 3 + 0];
        cap += hv[1] * larger_w[hid * 3 + 1];
        cap += hv[2] * larger_w[hid * 3 + 2];
        cap *= glg[hid];
        hbf[(size_t)row * 768 + hid] = f2bf(x[(size_t)row * 768 + hid] + cap);
    }
}

// ---------------------------------------------------------------------------
// Launcher
// ---------------------------------------------------------------------------
extern "C" void kernel_launch(void* const* d_in, const int* in_sizes, int n_in,
                              void* d_out, int out_size, void* d_ws, size_t ws_size,
                              hipStream_t stream)
{
    const float* x        = (const float*)d_in[0];
    const int*   tptr     = (const int*)  d_in[1];
    const float* sptr     = (const float*)d_in[2];
    const float* fc1_b    = (const float*)d_in[4];
    const float* fc2_b    = (const float*)d_in[6];
    const float* efc1     = (const float*)d_in[7];
    const float* efc2     = (const float*)d_in[8];
    const float* sfc1_w   = (const float*)d_in[9];
    const float* sfc1_b   = (const float*)d_in[10];
    const float* sfc2_w   = (const float*)d_in[11];
    const float* sfc2_b   = (const float*)d_in[12];
    const float* rw       = (const float*)d_in[13];
    const float* larger_w = (const float*)d_in[14];
    const float* larger_b = (const float*)d_in[15];
    const float* elarger  = (const float*)d_in[16];
    float* out = (float*)d_out;

    char* ws = (char*)d_ws;
    size_t off = 0;
    auto carve = [&](size_t bytes) { size_t o = off; off = (off + bytes + 255) & ~(size_t)255; return o; };

    unsigned short* xbf   = (unsigned short*)(ws + carve(16384L * 768 * 2)); // also reused as h
    unsigned short* fc1w  = (unsigned short*)(ws + carve(2048L * 768 * 2));
    unsigned short* fc2w  = (unsigned short*)(ws + carve(768L * 2048 * 2));
    unsigned short* wc    = (unsigned short*)(ws + carve(128L * 768 * 2));
    float*          bc    = (float*)(ws + carve(128 * 4));
    float*          gfc1  = (float*)(ws + carve(2048 * 4));
    float*          gfc2  = (float*)(ws + carve(768 * 4));
    float*          glg   = (float*)(ws + carve(768 * 4));
    unsigned short* rwt   = (unsigned short*)(ws + carve(30L * 128 * 384 * 2));
    float*          semR  = (float*)(ws + carve(16384L * 32 * 4));
    unsigned short* semS  = (unsigned short*)(ws + carve(128L * 10 * 384 * 2));
    float*          prio  = (float*)(ws + carve(3L * 128 * 10 * 128 * 4));
    float*          vote  = (float*)(ws + carve(49152L * 4));
    unsigned short* a1    = (unsigned short*)(ws + carve(16384L * 2048 * 2));
    (void)ws_size; (void)n_in; (void)in_sizes; (void)out_size;

    // --- prep: conversions, gates, fused capsule weights ---
    cvt_bf16_kernel<<<2048, 256, 0, stream>>>(x, xbf, 16384L * 768);
    cvt_bf16_kernel<<<2048, 256, 0, stream>>>((const float*)d_in[3], fc1w, 2048L * 768);
    cvt_bf16_kernel<<<2048, 256, 0, stream>>>((const float*)d_in[5], fc2w, 768L * 2048);
    transpose_rw_kernel<<<2048, 256, 0, stream>>>(rw, rwt);
    fuse_wc_kernel<<<(128 * 768 + 255) / 256, 256, 0, stream>>>(sfc1_w, sfc1_b, sfc2_w, sfc2_b, wc, bc);
    gates_kernel<<<(2048 + 255) / 256, 256, 0, stream>>>(tptr, sptr, efc1, efc2, elarger, gfc1, gfc2, glg);

    // --- semantic capsules (fused 2-layer linear): [16384x768] x [768x30pad128] ---
    gemm_kernel<true, false, false, false, false><<<dim3(1, 128), 256, 0, stream>>>(
        xbf, 768, wc, 768, 768, semR, nullptr, 32, 30, bc, nullptr, nullptr);

    // --- squash over tasks -> bf16 [b][n][384] ---
    squash_tasks_kernel<<<(128 * 128 * 3 + 255) / 256, 256, 0, stream>>>(semR, semS);

    // --- priors: 30 batched 128x128x384 WMMA GEMMs ---
    priors_kernel<<<dim3(1, 30), 256, 0, stream>>>(semS, rwt, prio);

    // --- dynamic routing-by-agreement (3 iters) ---
    routing_kernel<<<384, 128, 0, stream>>>(prio, tptr, vote);

    // --- capsule output + residual into bf16 h (reuses xbf buffer) ---
    build_h_kernel<<<16384, 256, 0, stream>>>(x, vote, larger_w, larger_b, glg, xbf);

    // --- adapter fc1: a1 = relu(h @ fc1^T + b1) * gfc1, stored bf16 ---
    gemm_kernel<true, true, true, false, true><<<dim3(16, 128), 256, 0, stream>>>(
        xbf, 768, fc1w, 768, 768, nullptr, a1, 2048, 2048, fc1_b, gfc1, nullptr);

    // --- adapter fc2 + residual: out = x + relu(a1 @ fc2^T + b2) * gfc2 ---
    gemm_kernel<true, true, true, true, false><<<dim3(6, 128), 256, 0, stream>>>(
        a1, 2048, fc2w, 2048, 2048, out, nullptr, 768, 768, fc2_b, gfc2, x);
}
